// MultiHeadAttention_37606733644076
// MI455X (gfx1250) — matrix operand order repaired and checked by reading, not compiled, not run on hardware
//
#include <hip/hip_runtime.h>
#include <hip/hip_bf16.h>
#include <stdint.h>

#define D_EMB    1024
#define NHEADS   16
#define DHEAD    64
#define BATCH    8
#define SEQ      1024
#define MROWS    (BATCH*SEQ)    // 8192
#define QKV_COLS (3*D_EMB)      // 3072

typedef __attribute__((ext_vector_type(16))) __bf16 v16bf;
typedef __attribute__((ext_vector_type(8)))  float  v8f;
typedef __attribute__((ext_vector_type(4)))  int    v4i;

union Frag {
    v16bf v;
    uint4 u[2];
};

// scheduler directives: 0x100 = DS read group, 0x008 = MFMA/WMMA group
#define SCHED_DS_READ(n) __builtin_amdgcn_sched_group_barrier(0x100, (n), 0)
#define SCHED_WMMA(n)    __builtin_amdgcn_sched_group_barrier(0x008, (n), 0)

// pin "DS x4 lead, then (WMMA, DS x2) x (nw-2), then WMMA x2" pipeline shape
__device__ __forceinline__ void sched_pipeline(int nw) {
    SCHED_DS_READ(4);
    for (int t = 0; t < nw - 2; ++t) { SCHED_WMMA(1); SCHED_DS_READ(2); }
    SCHED_WMMA(2);
}

__device__ __forceinline__ __bf16 f2bf(float f) {
    union { float f; uint32_t u; } x; x.f = f;
    uint32_t r = x.u + 0x7FFFu + ((x.u >> 16) & 1u);   // round-to-nearest-even
    union { uint16_t s; __bf16 b; } y; y.s = (uint16_t)(r >> 16);
    return y.b;
}

__device__ __forceinline__ v8f wmma_bf16(const Frag& a, const Frag& b, v8f c) {
    return __builtin_amdgcn_wmma_f32_16x16x32_bf16(
        false, a.v, false, b.v, (short)0, c, false, false);
}

__device__ __forceinline__ float rmax16(float v) {
    #pragma unroll
    for (int i = 1; i < 16; i <<= 1) v = fmaxf(v, __shfl_xor(v, i, 32));
    return v;
}
__device__ __forceinline__ float rsum16(float v) {
    #pragma unroll
    for (int i = 1; i < 16; i <<= 1) v += __shfl_xor(v, i, 32);
    return v;
}

// ---- async global->LDS copy (CDNA5 GLOBAL_LOAD_ASYNC_TO_LDS_B128, ASYNCcnt) ----
#if defined(__has_builtin)
#if __has_builtin(__builtin_amdgcn_global_load_async_to_lds_b128)
#define HAVE_ASYNC 1
#else
#define HAVE_ASYNC 0
#endif
#else
#define HAVE_ASYNC 0
#endif

__device__ __forceinline__ void cp16_g2l(__bf16* ldst, const __bf16* gsrc) {
#if HAVE_ASYNC
    __builtin_amdgcn_global_load_async_to_lds_b128(
        (__attribute__((address_space(1))) v4i*)gsrc,
        (__attribute__((address_space(3))) v4i*)ldst, 0, 0);
#else
    *(uint4*)ldst = *(const uint4*)gsrc;
#endif
}

__device__ __forceinline__ void wait_stage_then_sync() {
#if HAVE_ASYNC
#if __has_builtin(__builtin_amdgcn_s_wait_asynccnt)
    __builtin_amdgcn_s_wait_asynccnt(0);
#else
    asm volatile("s_wait_asynccnt 0x0" ::: "memory");
#endif
#endif
    __syncthreads();
}

// ---------------------------------------------------------------- prep kernels
__global__ void cvt_f32_bf16(const float* __restrict__ in, __bf16* __restrict__ out, int n) {
    int i = blockIdx.x * blockDim.x + threadIdx.x;
    if (i < n) out[i] = f2bf(in[i]);
}

// in: [rows][cols] f32  ->  out: [cols][rows] bf16
__global__ void transpose_cvt(const float* __restrict__ in, __bf16* __restrict__ out,
                              int rows, int cols) {
    int i = blockIdx.x * blockDim.x + threadIdx.x;
    if (i < rows * cols) {
        int r = i / cols, c = i % cols;
        out[(size_t)c * rows + r] = f2bf(in[i]);
    }
}

// ================================================================ GEMM core
// Block: 256 threads = 8 waves. Tile: 128 rows x 128 cols. K-step 32.
// B tile [128 cols][32 k] staged in LDS (double buffered, async), shared by all waves.

__device__ __forceinline__ void stage_b(__bf16 (*bsh)[32], const __bf16* __restrict__ wt,
                                        int col0, int kb) {
    const int tid = threadIdx.x;
    #pragma unroll
    for (int i = 0; i < 2; ++i) {
        int c  = tid + i * 256;        // 512 chunks of 8 bf16
        int n  = c >> 2;
        int ko = (c & 3) * 8;
        cp16_g2l(&bsh[n][ko], wt + (size_t)(col0 + n) * D_EMB + kb + ko);
    }
}

__device__ __forceinline__ void load_a(Frag& a, const __bf16* __restrict__ src,
                                       int row, int kb, int grp) {
    const __bf16* ap = src + (size_t)row * D_EMB + kb + grp * 8;
    a.u[0] = *(const uint4*)(ap);
    a.u[1] = *(const uint4*)(ap + 16);
}

__device__ __forceinline__ void load_b_frag(Frag& b, const __bf16 (*bsh)[32],
                                            int t, int lr, int grp) {
    const __bf16* bp = &bsh[t * 16 + lr][grp * 8];
    b.u[0] = *(const uint4*)(bp);
    b.u[1] = *(const uint4*)(bp + 16);
}

// ---------------------------------------------------------------- QKV GEMM
__global__ __launch_bounds__(256) void gemm_qkv(
    const __bf16* __restrict__ xb,     // [8192][1024]
    const __bf16* __restrict__ wt,     // [3072][1024] (transposed weights)
    const float*  __restrict__ bias,   // [3072]
    __bf16* __restrict__ qbuf,         // [B*H][N][64]
    __bf16* __restrict__ kbuf,         // [B*H][N][64]
    __bf16* __restrict__ vtbuf)        // [B*H][64][N]
{
    __shared__ __bf16 bsh[2][128][32];
    const int lane = threadIdx.x & 31;
    const int wave = threadIdx.x >> 5;
    const int lr   = lane & 15;
    const int grp  = lane >> 4;
    const int row0 = blockIdx.x * 128 + wave * 16;
    const int col0 = blockIdx.y * 128;

    v8f acc[8] = {};
    Frag a[2];

    stage_b(bsh[0], wt, col0, 0);
    load_a(a[0], xb, row0 + lr, 0, grp);
    wait_stage_then_sync();

    for (int kb = 0; kb < D_EMB; kb += 32) {
        const int cur = (kb >> 5) & 1;
        const bool more = (kb + 32) < D_EMB;
        if (more) {
            stage_b(bsh[cur ^ 1], wt, col0, kb + 32);
            load_a(a[cur ^ 1], xb, row0 + lr, kb + 32, grp);
        }
        Frag bf[2];
        load_b_frag(bf[0], bsh[cur], 0, lr, grp);
        #pragma unroll
        for (int t = 0; t < 8; ++t) {
            if (t < 7) load_b_frag(bf[(t + 1) & 1], bsh[cur], t + 1, lr, grp);
            acc[t] = wmma_bf16(a[cur], bf[t & 1], acc[t]);
        }
        sched_pipeline(8);
        if (more) wait_stage_then_sync();
    }

    // epilogue: decode (h, d, qkv) interleave and scatter
    #pragma unroll
    for (int t = 0; t < 8; ++t) {
        #pragma unroll
        for (int j = 0; j < 8; ++j) {
            int m = row0 + j + grp * 8;
            int c = col0 + t * 16 + lr;
            float val = acc[t][j] + bias[c];
            int s  = c % 3;
            int h  = c / (DHEAD * 3);
            int di = (c % (DHEAD * 3)) / 3;
            int bb = m >> 10;
            int n  = m & 1023;
            __bf16 bv = f2bf(val);
            size_t bh = (size_t)bb * NHEADS + h;
            if (s == 0)      qbuf [(bh * SEQ + n) * DHEAD + di] = bv;
            else if (s == 1) kbuf [(bh * SEQ + n) * DHEAD + di] = bv;
            else             vtbuf[(bh * DHEAD + di) * SEQ + n] = bv;
        }
    }
}

// ---------------------------------------------------------------- projection GEMM
__global__ __launch_bounds__(256) void gemm_proj(
    const __bf16* __restrict__ ab,     // [8192][1024]
    const __bf16* __restrict__ wt,     // [1024][1024] (transposed)
    const float*  __restrict__ bias,   // [1024]
    float* __restrict__ out)           // [8192][1024]
{
    __shared__ __bf16 bsh[2][128][32];
    const int lane = threadIdx.x & 31;
    const int wave = threadIdx.x >> 5;
    const int lr   = lane & 15;
    const int grp  = lane >> 4;
    const int row0 = blockIdx.x * 128 + wave * 16;
    const int col0 = blockIdx.y * 128;

    v8f acc[8] = {};
    Frag a[2];

    stage_b(bsh[0], wt, col0, 0);
    load_a(a[0], ab, row0 + lr, 0, grp);
    wait_stage_then_sync();

    for (int kb = 0; kb < D_EMB; kb += 32) {
        const int cur = (kb >> 5) & 1;
        const bool more = (kb + 32) < D_EMB;
        if (more) {
            stage_b(bsh[cur ^ 1], wt, col0, kb + 32);
            load_a(a[cur ^ 1], ab, row0 + lr, kb + 32, grp);
        }
        Frag bf[2];
        load_b_frag(bf[0], bsh[cur], 0, lr, grp);
        #pragma unroll
        for (int t = 0; t < 8; ++t) {
            if (t < 7) load_b_frag(bf[(t + 1) & 1], bsh[cur], t + 1, lr, grp);
            acc[t] = wmma_bf16(a[cur], bf[t & 1], acc[t]);
        }
        sched_pipeline(8);
        if (more) wait_stage_then_sync();
    }

    #pragma unroll
    for (int t = 0; t < 8; ++t) {
        #pragma unroll
        for (int j = 0; j < 8; ++j) {
            int m = row0 + j + grp * 8;
            int c = col0 + t * 16 + lr;
            out[(size_t)m * D_EMB + c] = acc[t][j] + bias[c];
        }
    }
}

// ---------------------------------------------------------------- flash attention
// Block: 128 threads = 4 waves, all share key-block kb. K tile [32 keys][64 d] and
// V tile [64 d][32 keys] staged in LDS, double buffered; Q resident in registers.
__global__ __launch_bounds__(128) void attn(
    const __bf16* __restrict__ qbuf,
    const __bf16* __restrict__ kbuf,
    const __bf16* __restrict__ vtbuf,
    __bf16* __restrict__ attout)       // [B][N][H*64]
{
    __shared__ __bf16 ksh[2][32][64];
    __shared__ __bf16 vsh[2][64][32];
    __shared__ __bf16 plds[4][16][32];

    const int tid  = threadIdx.x;
    const int lane = tid & 31;
    const int wave = tid >> 5;
    const int lr   = lane & 15;
    const int grp  = lane >> 4;
    const int bh   = blockIdx.x;               // b*16 + h
    const int q0   = blockIdx.y * 64 + wave * 16;

    const __bf16* qb  = qbuf  + (size_t)bh * SEQ * DHEAD;
    const __bf16* kbp = kbuf  + (size_t)bh * SEQ * DHEAD;
    const __bf16* vtp = vtbuf + (size_t)bh * DHEAD * SEQ;

    Frag qa[2];
    #pragma unroll
    for (int kd = 0; kd < 2; ++kd) {
        const __bf16* ap = qb + (size_t)(q0 + lr) * DHEAD + kd * 32 + grp * 8;
        qa[kd].u[0] = *(const uint4*)(ap);
        qa[kd].u[1] = *(const uint4*)(ap + 16);
    }

    v8f o[4] = {};
    float mx[8], sm[8];
    #pragma unroll
    for (int j = 0; j < 8; ++j) { mx[j] = -3.0e38f; sm[j] = 0.f; }

    // stage key-block kb into buffer buf
    auto stage_kv = [&](int buf, int kb) {
        #pragma unroll
        for (int i = 0; i < 2; ++i) {
            int c  = tid + i * 128;             // K: 256 chunks of 8 bf16
            int n  = c >> 3;
            int ko = (c & 7) * 8;
            cp16_g2l(&ksh[buf][n][ko], kbp + (size_t)(kb + n) * DHEAD + ko);
        }
        #pragma unroll
        for (int i = 0; i < 2; ++i) {
            int c  = tid + i * 128;             // V: 256 chunks of 8 bf16
            int di = c >> 2;
            int ko = (c & 3) * 8;
            cp16_g2l(&vsh[buf][di][ko], vtp + (size_t)di * SEQ + kb + ko);
        }
    };

    stage_kv(0, 0);
    wait_stage_then_sync();

    for (int kb = 0; kb < SEQ; kb += 32) {
        const int cur = (kb >> 5) & 1;
        const bool more = (kb + 32) < SEQ;
        if (more) stage_kv(cur ^ 1, kb + 32);

        // ---- S = Q K^T : rolling 2-deep K-fragment pipeline ----
        auto load_k_frag = [&](Frag& b, int tile, int kd) {
            const __bf16* kp = &ksh[cur][tile * 16 + lr][kd * 32 + grp * 8];
            b.u[0] = *(const uint4*)(kp);
            b.u[1] = *(const uint4*)(kp + 16);
        };
        v8f s0 = {}, s1 = {};
        {
            const int seq_t[4] = {0, 1, 0, 1};
            const int seq_k[4] = {0, 0, 1, 1};
            Frag kf[2];
            load_k_frag(kf[0], 0, 0);
            #pragma unroll
            for (int i = 0; i < 4; ++i) {
                if (i < 3) load_k_frag(kf[(i + 1) & 1], seq_t[i + 1], seq_k[i + 1]);
                if (seq_t[i] == 0) s0 = wmma_bf16(qa[seq_k[i]], kf[i & 1], s0);
                else               s1 = wmma_bf16(qa[seq_k[i]], kf[i & 1], s1);
            }
            sched_pipeline(4);
        }

        // ---- online softmax over this 32-key block ----
        #pragma unroll
        for (int j = 0; j < 8; ++j) {
            float rm   = rmax16(fmaxf(s0[j], s1[j]));
            float mnew = fmaxf(mx[j], rm);
            float scl  = __expf(mx[j] - mnew);
            float p0   = __expf(s0[j] - mnew);
            float p1   = __expf(s1[j] - mnew);
            float rs   = rsum16(p0 + p1);
            sm[j] = sm[j] * scl + rs;
            mx[j] = mnew;
            #pragma unroll
            for (int t = 0; t < 4; ++t) o[t][j] *= scl;
            plds[wave][j + grp * 8][lr]      = f2bf(p0);
            plds[wave][j + grp * 8][16 + lr] = f2bf(p1);
        }
        __builtin_amdgcn_wave_barrier();   // per-wave LDS region; HW keeps DS in order
        Frag pf;
        const __bf16* pp = &plds[wave][lr][grp * 8];
        pf.u[0] = *(const uint4*)(pp);
        pf.u[1] = *(const uint4*)(pp + 16);
        __builtin_amdgcn_wave_barrier();

        // ---- O += P V : rolling 2-deep V-fragment pipeline ----
        auto load_v_frag = [&](Frag& b, int t) {
            const __bf16* vp = &vsh[cur][t * 16 + lr][grp * 8];
            b.u[0] = *(const uint4*)(vp);
            b.u[1] = *(const uint4*)(vp + 16);
        };
        {
            Frag vf[2];
            load_v_frag(vf[0], 0);
            #pragma unroll
            for (int t = 0; t < 4; ++t) {
                if (t < 3) load_v_frag(vf[(t + 1) & 1], t + 1);
                o[t] = wmma_bf16(pf, vf[t & 1], o[t]);
            }
            sched_pipeline(4);
        }
        if (more) wait_stage_then_sync();
    }

    const int b = bh >> 4, h = bh & 15;
    #pragma unroll
    for (int j = 0; j < 8; ++j) {
        float inv = 1.0f / (sm[j] * 32.0f);   // 1/rowsum * 1/sqrt(1024)
        int nq = q0 + j + grp * 8;
        #pragma unroll
        for (int t = 0; t < 4; ++t) {
            int di = t * 16 + lr;
            attout[((size_t)(b * SEQ + nq)) * D_EMB + h * DHEAD + di] = f2bf(o[t][j] * inv);
        }
    }
}

// ---------------------------------------------------------------- launch
extern "C" void kernel_launch(void* const* d_in, const int* in_sizes, int n_in,
                              void* d_out, int out_size, void* d_ws, size_t ws_size,
                              hipStream_t stream) {
    (void)in_sizes; (void)n_in; (void)out_size; (void)ws_size;
    const float* x      = (const float*)d_in[0];
    const float* w_qkv  = (const float*)d_in[1];
    const float* b_qkv  = (const float*)d_in[2];
    const float* w_proj = (const float*)d_in[3];
    const float* b_proj = (const float*)d_in[4];
    float* out = (float*)d_out;

    char* ws = (char*)d_ws;
    __bf16* xb     = (__bf16*)ws;  ws += (size_t)MROWS * D_EMB * 2;
    __bf16* wqkvt  = (__bf16*)ws;  ws += (size_t)QKV_COLS * D_EMB * 2;
    __bf16* wprojt = (__bf16*)ws;  ws += (size_t)D_EMB * D_EMB * 2;
    __bf16* qbuf   = (__bf16*)ws;  ws += (size_t)BATCH * NHEADS * SEQ * DHEAD * 2;
    __bf16* kbuf   = (__bf16*)ws;  ws += (size_t)BATCH * NHEADS * SEQ * DHEAD * 2;
    __bf16* vtbuf  = (__bf16*)ws;  ws += (size_t)BATCH * NHEADS * SEQ * DHEAD * 2;
    __bf16* attb   = (__bf16*)ws;  ws += (size_t)MROWS * D_EMB * 2;

    cvt_f32_bf16<<<(MROWS * D_EMB + 255) / 256, 256, 0, stream>>>(x, xb, MROWS * D_EMB);
    transpose_cvt<<<(D_EMB * QKV_COLS + 255) / 256, 256, 0, stream>>>(w_qkv, wqkvt, D_EMB, QKV_COLS);
    transpose_cvt<<<(D_EMB * D_EMB + 255) / 256, 256, 0, stream>>>(w_proj, wprojt, D_EMB, D_EMB);

    gemm_qkv<<<dim3(MROWS / 128, QKV_COLS / 128), 256, 0, stream>>>(
        xb, wqkvt, b_qkv, qbuf, kbuf, vtbuf);

    attn<<<dim3(BATCH * NHEADS, SEQ / 64), 128, 0, stream>>>(qbuf, kbuf, vtbuf, attb);

    gemm_proj<<<dim3(MROWS / 128, D_EMB / 128), 256, 0, stream>>>(attb, wprojt, b_proj, out);
}
